// DotProductAttention_71339406786914
// MI455X (gfx1250) — matrix operand order repaired and checked by reading, not compiled, not run on hardware
//
#include <hip/hip_runtime.h>

typedef __attribute__((ext_vector_type(16))) __bf16 v16bf;
typedef __attribute__((ext_vector_type(8)))  __bf16 v8bf;
typedef __attribute__((ext_vector_type(8)))  float  v8f;

#define SEQ    1024   // n == m
#define DHEAD  128    // d == dv
#define QTILE  128    // q rows per block (8 waves x 16 rows)
#define KTA    64     // keys per iteration, fast path (2 x K32 for PV WMMA)
#define KTF    32     // keys per iteration, fallback
#define SCALE  0.0883883476483184405501f  // 1/sqrt(128)

__device__ __forceinline__ unsigned short f2bfu(float f) {
  return __builtin_bit_cast(unsigned short, (__bf16)f);   // hardware f32->bf16 cvt
}

__device__ __forceinline__ v16bf cat8(v8bf lo, v8bf hi) {
  return __builtin_shufflevector(lo, hi, 0,1,2,3,4,5,6,7,8,9,10,11,12,13,14,15);
}

// DPP row_xmask butterfly step (stays within a 16-lane row; pure VALU, no DS).
template <int CTRL>
__device__ __forceinline__ float dppx(float x) {
  return __builtin_bit_cast(float,
      __builtin_amdgcn_update_dpp(0, __builtin_bit_cast(int, x), CTRL, 0xf, 0xf, true));
}
__device__ __forceinline__ float rowmax16(float x) {
  x = fmaxf(x, dppx<0x161>(x));   // row_xmask:1
  x = fmaxf(x, dppx<0x162>(x));   // row_xmask:2
  x = fmaxf(x, dppx<0x164>(x));   // row_xmask:4
  x = fmaxf(x, dppx<0x168>(x));   // row_xmask:8
  return x;
}
__device__ __forceinline__ float rowsum16(float x) {
  x += dppx<0x161>(x);
  x += dppx<0x162>(x);
  x += dppx<0x164>(x);
  x += dppx<0x168>(x);
  return x;
}

// ---------------------------------------------------------------------------
// Prep kernel: K (f32, [B][m][d]) -> Kb (bf16, same layout)
//              V (f32, [B][m][dv]) -> Vt (bf16, [B][dv][m] transposed)
// ---------------------------------------------------------------------------
__global__ void cvt_kv_kernel(const float* __restrict__ K, const float* __restrict__ V,
                              unsigned short* __restrict__ Kb, unsigned short* __restrict__ Vt) {
  size_t f = (size_t)blockIdx.x * 256 + threadIdx.x;
  Kb[f] = f2bfu(K[f]);
  size_t b   = f >> 17;            // SEQ*DHEAD = 131072
  int    rem = (int)(f & 131071);
  int    k   = rem >> 7;
  int    dd  = rem & 127;
  Vt[(b << 17) + ((size_t)dd << 10) + k] = f2bfu(V[f]);
}

// ---------------------------------------------------------------------------
// Fast path: bf16 K/V from workspace, 64-key tiles, double-buffered async LDS.
// ---------------------------------------------------------------------------
__global__ __launch_bounds__(256, 1)
void fa_fwd_async(const float* __restrict__ Qg, const unsigned short* __restrict__ Kbg,
                  const unsigned short* __restrict__ Vtg, const int* __restrict__ VLg,
                  float* __restrict__ Og) {
  __shared__ alignas(16) unsigned short sK [2][KTA][DHEAD];   // bf16 [key][d]    2x16KB
  __shared__ alignas(16) unsigned short sVT[2][DHEAD][KTA];   // bf16 [dv][key]   2x16KB
  __shared__ alignas(16) unsigned short sP [8][16][KTA];      // per-wave P slab  16KB

  const int tid  = threadIdx.x;
  const int wave = tid >> 5;
  const int lane = tid & 31;
  const int g    = lane >> 4;
  const int n16  = lane & 15;

  const int b  = blockIdx.y;
  const int q0 = blockIdx.x * QTILE;

  const float*          Q   = Qg  + ((size_t)b * SEQ + q0) * DHEAD;
  const unsigned short* Kbb = Kbg + ((size_t)b << 17);
  const unsigned short* Vtb = Vtg + ((size_t)b << 17);
  float*                O   = Og  + ((size_t)b * SEQ + q0) * DHEAD;
  const int vl = VLg[b];

  // ---- Q A-fragments, pre-scaled by 1/sqrt(d).
  v16bf aQ[4];
  {
    const float* qrow = Q + (size_t)(wave * 16 + n16) * DHEAD;
    #pragma unroll
    for (int c = 0; c < 4; ++c) {
      const float4* p0 = (const float4*)(qrow + c * 32 + 8 * g);
      const float4* p1 = (const float4*)(qrow + c * 32 + 16 + 8 * g);
      float4 x0 = p0[0], x1 = p0[1], y0 = p1[0], y1 = p1[1];
      unsigned short hv[16] = {
        f2bfu(x0.x*SCALE), f2bfu(x0.y*SCALE), f2bfu(x0.z*SCALE), f2bfu(x0.w*SCALE),
        f2bfu(x1.x*SCALE), f2bfu(x1.y*SCALE), f2bfu(x1.z*SCALE), f2bfu(x1.w*SCALE),
        f2bfu(y0.x*SCALE), f2bfu(y0.y*SCALE), f2bfu(y0.z*SCALE), f2bfu(y0.w*SCALE),
        f2bfu(y1.x*SCALE), f2bfu(y1.y*SCALE), f2bfu(y1.z*SCALE), f2bfu(y1.w*SCALE) };
      v16bf a;
      #pragma unroll
      for (int i = 0; i < 16; ++i) a[i] = __builtin_bit_cast(__bf16, hv[i]);
      aQ[c] = a;
    }
  }

  // ---- async tile staging (ASYNCcnt): pure byte copies of pre-converted bf16
  auto stage = [&](int buf, int kb) {
    unsigned           ldsK = (unsigned)(uintptr_t)&sK[buf][0][0];
    unsigned long long srcK = (unsigned long long)(uintptr_t)(Kbb + (size_t)kb * DHEAD);
    #pragma unroll
    for (int i = 0; i < 4; ++i) {                 // 16KB contiguous: 1024 x 16B chunks
      int off = (tid + 256 * i) * 16;
      asm volatile("global_load_async_to_lds_b128 %0, %1, off"
                   :: "v"(ldsK + off), "v"(srcK + (unsigned long long)off) : "memory");
    }
    unsigned           ldsV = (unsigned)(uintptr_t)&sVT[buf][0][0];
    unsigned long long srcV = (unsigned long long)(uintptr_t)(Vtb + kb);
    #pragma unroll
    for (int i = 0; i < 4; ++i) {                 // 128 rows x 128B (8 x 16B chunks)
      int c    = tid + 256 * i;
      int row  = c >> 3;
      int part = c & 7;
      asm volatile("global_load_async_to_lds_b128 %0, %1, off"
                   :: "v"(ldsV + row * 128 + part * 16),
                      "v"(srcV + (unsigned long long)(row * 2048 + part * 16)) : "memory");
    }
  };

  v8f acc[8] = {};
  float rowm[8], rowl[8];
  #pragma unroll
  for (int e = 0; e < 8; ++e) { rowm[e] = -__builtin_inff(); rowl[e] = 0.0f; }

  stage(0, 0);

  for (int kb = 0; kb < SEQ; kb += KTA) {
    const int cur = (kb >> 6) & 1;
    asm volatile("s_wait_asynccnt 0x0" ::: "memory");
    __syncthreads();
    if (kb + KTA < SEQ) stage(cur ^ 1, kb + KTA);   // prefetch overlaps WMMAs below

    // ---- S = (Q*scale) * K^T : four 16x16 key tiles, 4 K32 chunks each
    v8f s[4] = {};
    #pragma unroll
    for (int h = 0; h < 4; ++h) {
      #pragma unroll
      for (int c = 0; c < 4; ++c) {
        v8bf lo = *(const v8bf*)&sK[cur][16 * h + n16][c * 32 + 8 * g];
        v8bf hi = *(const v8bf*)&sK[cur][16 * h + n16][c * 32 + 16 + 8 * g];
        s[h] = __builtin_amdgcn_wmma_f32_16x16x32_bf16(
                   false, aQ[c], false, cat8(lo, hi), (short)0, s[h], false, false);
      }
    }

    // ---- online softmax (C/D layout: row = e + 8g, col = lane%16)
    bool msk[4];
    #pragma unroll
    for (int h = 0; h < 4; ++h) msk[h] = (kb + 16 * h + n16) >= vl;
    #pragma unroll
    for (int e = 0; e < 8; ++e) {
      float sv[4];
      #pragma unroll
      for (int h = 0; h < 4; ++h) sv[h] = msk[h] ? -10000000.0f : s[h][e];
      float mx = fmaxf(fmaxf(sv[0], sv[1]), fmaxf(sv[2], sv[3]));
      mx = rowmax16(mx);
      float mo = rowm[e];
      float mn = fmaxf(mo, mx);
      rowm[e] = mn;
      float cf = __expf(mo - mn);
      float ps = 0.0f;
      #pragma unroll
      for (int h = 0; h < 4; ++h) {
        float p = __expf(sv[h] - mn);
        ps += p;
        sP[wave][e + 8 * g][16 * h + n16] = f2bfu(p);
      }
      ps = rowsum16(ps);
      rowl[e] = rowl[e] * cf + ps;
      #pragma unroll
      for (int t = 0; t < 8; ++t) acc[t][e] *= cf;
    }

    // ---- re-stripe P via per-wave LDS slab (DS ops in-order per wave)
    v16bf aP[2];
    #pragma unroll
    for (int kc = 0; kc < 2; ++kc) {
      v8bf plo = *(const v8bf*)&sP[wave][n16][kc * 32 + 8 * g];
      v8bf phi = *(const v8bf*)&sP[wave][n16][kc * 32 + 16 + 8 * g];
      aP[kc] = cat8(plo, phi);
    }

    // ---- O += P * V : 8 dv tiles x 2 K32 chunks
    #pragma unroll
    for (int t = 0; t < 8; ++t) {
      #pragma unroll
      for (int kc = 0; kc < 2; ++kc) {
        v8bf vlo = *(const v8bf*)&sVT[cur][t * 16 + n16][kc * 32 + 8 * g];
        v8bf vhi = *(const v8bf*)&sVT[cur][t * 16 + n16][kc * 32 + 16 + 8 * g];
        acc[t] = __builtin_amdgcn_wmma_f32_16x16x32_bf16(
                     false, aP[kc], false, cat8(vlo, vhi), (short)0, acc[t], false, false);
      }
    }
  }

  #pragma unroll
  for (int e = 0; e < 8; ++e) {
    float inv = 1.0f / rowl[e];
    int row = wave * 16 + e + 8 * g;
    #pragma unroll
    for (int t = 0; t < 8; ++t) {
      O[(size_t)row * DHEAD + t * 16 + n16] = acc[t][e] * inv;
    }
  }
}

// ---------------------------------------------------------------------------
// Fallback (workspace too small): in-kernel f32->bf16 staging, 32-key tiles.
// ---------------------------------------------------------------------------
__global__ __launch_bounds__(256, 1)
void fa_fwd_fallback(const float* __restrict__ Qg, const float* __restrict__ Kg,
                     const float* __restrict__ Vg, const int* __restrict__ VLg,
                     float* __restrict__ Og) {
  __shared__ alignas(16) unsigned short sK [KTF][DHEAD];
  __shared__ alignas(16) unsigned short sVT[DHEAD][KTF];
  __shared__ alignas(16) unsigned short sP [8][16][KTF];

  const int tid  = threadIdx.x;
  const int wave = tid >> 5;
  const int lane = tid & 31;
  const int g    = lane >> 4;
  const int n16  = lane & 15;

  const int b  = blockIdx.y;
  const int q0 = blockIdx.x * QTILE;

  const float* Q = Qg + ((size_t)b * SEQ + q0) * DHEAD;
  const float* K = Kg + (size_t)b * SEQ * DHEAD;
  const float* V = Vg + (size_t)b * SEQ * DHEAD;
  float*       O = Og + ((size_t)b * SEQ + q0) * DHEAD;
  const int vl = VLg[b];

  v16bf aQ[4];
  {
    const float* qrow = Q + (size_t)(wave * 16 + n16) * DHEAD;
    #pragma unroll
    for (int c = 0; c < 4; ++c) {
      const float4* p0 = (const float4*)(qrow + c * 32 + 8 * g);
      const float4* p1 = (const float4*)(qrow + c * 32 + 16 + 8 * g);
      float4 x0 = p0[0], x1 = p0[1], y0 = p1[0], y1 = p1[1];
      unsigned short hv[16] = {
        f2bfu(x0.x*SCALE), f2bfu(x0.y*SCALE), f2bfu(x0.z*SCALE), f2bfu(x0.w*SCALE),
        f2bfu(x1.x*SCALE), f2bfu(x1.y*SCALE), f2bfu(x1.z*SCALE), f2bfu(x1.w*SCALE),
        f2bfu(y0.x*SCALE), f2bfu(y0.y*SCALE), f2bfu(y0.z*SCALE), f2bfu(y0.w*SCALE),
        f2bfu(y1.x*SCALE), f2bfu(y1.y*SCALE), f2bfu(y1.z*SCALE), f2bfu(y1.w*SCALE) };
      v16bf a;
      #pragma unroll
      for (int i = 0; i < 16; ++i) a[i] = __builtin_bit_cast(__bf16, hv[i]);
      aQ[c] = a;
    }
  }

  v8f acc[8] = {};
  float rowm[8], rowl[8];
  #pragma unroll
  for (int e = 0; e < 8; ++e) { rowm[e] = -__builtin_inff(); rowl[e] = 0.0f; }

  for (int kb = 0; kb < SEQ; kb += KTF) {
    __syncthreads();
    #pragma unroll
    for (int i = 0; i < 4; ++i) {
      int f4 = tid + 256 * i;
      int r  = f4 >> 5;
      int c4 = f4 & 31;
      float4 kx = ((const float4*)(K + (size_t)(kb + r) * DHEAD))[c4];
      uint2 w;
      w.x = (unsigned)f2bfu(kx.x) | ((unsigned)f2bfu(kx.y) << 16);
      w.y = (unsigned)f2bfu(kx.z) | ((unsigned)f2bfu(kx.w) << 16);
      *(uint2*)&sK[r][c4 * 4] = w;
      float4 vx = ((const float4*)(V + (size_t)(kb + r) * DHEAD))[c4];
      sVT[c4 * 4 + 0][r] = f2bfu(vx.x);
      sVT[c4 * 4 + 1][r] = f2bfu(vx.y);
      sVT[c4 * 4 + 2][r] = f2bfu(vx.z);
      sVT[c4 * 4 + 3][r] = f2bfu(vx.w);
    }
    __syncthreads();

    v8f s[2] = {};
    #pragma unroll
    for (int h = 0; h < 2; ++h) {
      #pragma unroll
      for (int c = 0; c < 4; ++c) {
        v8bf lo = *(const v8bf*)&sK[16 * h + n16][c * 32 + 8 * g];
        v8bf hi = *(const v8bf*)&sK[16 * h + n16][c * 32 + 16 + 8 * g];
        s[h] = __builtin_amdgcn_wmma_f32_16x16x32_bf16(
                   false, aQ[c], false, cat8(lo, hi), (short)0, s[h], false, false);
      }
    }

    const bool msk0 = (kb + n16)      >= vl;
    const bool msk1 = (kb + 16 + n16) >= vl;
    #pragma unroll
    for (int e = 0; e < 8; ++e) {
      float s0 = msk0 ? -10000000.0f : s[0][e];
      float s1 = msk1 ? -10000000.0f : s[1][e];
      float mx = rowmax16(fmaxf(s0, s1));
      float mo = rowm[e];
      float mn = fmaxf(mo, mx);
      rowm[e] = mn;
      float cf = __expf(mo - mn);
      float p0 = __expf(s0 - mn);
      float p1 = __expf(s1 - mn);
      float ps = rowsum16(p0 + p1);
      rowl[e] = rowl[e] * cf + ps;
      sP[wave][e + 8 * g][n16]      = f2bfu(p0);
      sP[wave][e + 8 * g][16 + n16] = f2bfu(p1);
      #pragma unroll
      for (int t = 0; t < 8; ++t) acc[t][e] *= cf;
    }

    v8bf plo = *(const v8bf*)&sP[wave][n16][8 * g];
    v8bf phi = *(const v8bf*)&sP[wave][n16][16 + 8 * g];
    v16bf aP = cat8(plo, phi);

    #pragma unroll
    for (int t = 0; t < 8; ++t) {
      v8bf vlo = *(const v8bf*)&sVT[t * 16 + n16][8 * g];
      v8bf vhi = *(const v8bf*)&sVT[t * 16 + n16][16 + 8 * g];
      acc[t] = __builtin_amdgcn_wmma_f32_16x16x32_bf16(
                   false, aP, false, cat8(vlo, vhi), (short)0, acc[t], false, false);
    }
  }

  #pragma unroll
  for (int e = 0; e < 8; ++e) {
    float inv = 1.0f / rowl[e];
    int row = wave * 16 + e + 8 * g;
    #pragma unroll
    for (int t = 0; t < 8; ++t) {
      O[(size_t)row * DHEAD + t * 16 + n16] = acc[t][e] * inv;
    }
  }
}

extern "C" void kernel_launch(void* const* d_in, const int* in_sizes, int n_in,
                              void* d_out, int out_size, void* d_ws, size_t ws_size,
                              hipStream_t stream) {
  const float* q  = (const float*)d_in[0];
  const float* k  = (const float*)d_in[1];
  const float* v  = (const float*)d_in[2];
  const int*   vl = (const int*)d_in[3];
  float*       o  = (float*)d_out;
  const int B = in_sizes[3];
  const size_t elems = (size_t)B * SEQ * DHEAD;
  dim3 grid(SEQ / QTILE, B);
  if (ws_size >= elems * 2 * sizeof(unsigned short)) {
    unsigned short* Kb = (unsigned short*)d_ws;
    unsigned short* Vt = Kb + elems;
    cvt_kv_kernel<<<dim3((unsigned)(elems / 256)), dim3(256), 0, stream>>>(k, v, Kb, Vt);
    fa_fwd_async<<<grid, dim3(256), 0, stream>>>(q, Kb, Vt, vl, o);
  } else {
    fa_fwd_fallback<<<grid, dim3(256), 0, stream>>>(q, k, v, vl, o);
  }
}
